// MessagePassingGNN_2834678415534
// MI455X (gfx1250) — matrix operand order repaired and checked by reading, not compiled, not run on hardware
//
#include <hip/hip_runtime.h>
#include <hip/hip_fp16.h>
#include <cstdint>

// ---------------- problem constants (match reference) ----------------
#define N_NODES 50000
#define N_EDGES 800000
#define IN_DIM  64
#define OUT_DIM 8
#define HID     32
#define MLP_HID 64
#define STEPS   3

#define NW   8      // waves per block (256 threads, wave32)
#define LDT  80     // inter-layer LDS tile row stride in halfs (160B rows)

typedef __attribute__((ext_vector_type(16))) _Float16     v16h;
typedef __attribute__((ext_vector_type(8)))  float        v8f;
typedef __attribute__((ext_vector_type(4)))  unsigned int u32x4;
typedef __attribute__((ext_vector_type(8)))  int          i32x8;
typedef __attribute__((ext_vector_type(4)))  int          i32x4;

union V16 { u32x4 u[2]; v16h h; };

// ---------------- WMMA fragment helpers (CDNA5 wave32 layouts) ----------------
// A fragment (16xK f16): lane<16: M=lane, elems = K kb+kh+0..7 / kb+16+kh..+7 (kh=0)
//                        lane>=16: same rows with kh=8.  Both chunks are 16B contiguous.
__device__ inline v16h ld_a_gl(const _Float16* rowp, int kh) {   // from global row
  V16 t;
  t.u[0] = *(const u32x4*)(rowp + kh);
  t.u[1] = *(const u32x4*)(rowp + 16 + kh);
  return t.h;
}
__device__ inline v16h ld_a(const _Float16* tile, int ld, int kb, int lane) {  // from LDS tile
  const _Float16* row = tile + (lane & 15) * ld + kb + (((lane >> 4) << 3));
  V16 t;
  t.u[0] = *(const u32x4*)row;
  t.u[1] = *(const u32x4*)(row + 16);
  return t.h;
}

// B fragment from fragment-packed weights: wp[((k>>4)*ncols + n)*16 + (k&15)]
// lane<16: N=nb+lane, K kofs+0..7 / kofs+16..23 ; lane>=16: N=nb+lane-16, K kofs+8..15 / kofs+24..31 (two 16B chunks)
__device__ inline v16h ld_bp(const _Float16* wp, int ncols, int nb, int kofs, int lane) {
  int n = nb + (lane & 15);
  int g = kofs >> 4;
  int kh = (lane >> 4) << 3;
  V16 t;
  t.u[0] = *(const u32x4*)(wp + (((g * ncols) + n) << 4) + kh);
  t.u[1] = *(const u32x4*)(wp + ((((g + 1) * ncols) + n) << 4) + kh);
  return t.h;
}

__device__ inline v8f wmma_k64(v16h a0, v16h a1, const _Float16* wp, int ncols, int nb, int lane) {
  v8f c = {};
  c = __builtin_amdgcn_wmma_f32_16x16x32_f16(false, a0, false, ld_bp(wp, ncols, nb, 0,  lane), (short)0, c, false, false);
  c = __builtin_amdgcn_wmma_f32_16x16x32_f16(false, a1, false, ld_bp(wp, ncols, nb, 32, lane), (short)0, c, false, false);
  return c;
}
__device__ inline v8f wmma_k32(v16h a0, const _Float16* wp, int ncols, int nb, int lane) {
  v8f c = {};
  c = __builtin_amdgcn_wmma_f32_16x16x32_f16(false, a0, false, ld_bp(wp, ncols, nb, 0, lane), (short)0, c, false, false);
  return c;
}

// C/D layout: c[r] -> (M = r + 8*(lane>>4), N = nb + (lane&15))
__device__ inline void st_c_f16(_Float16* tile, int ld, int nb, int lane, v8f c, float bb, bool tanh_act) {
  int n  = nb + (lane & 15);
  int mb = (lane >> 4) << 3;
#pragma unroll
  for (int r = 0; r < 8; ++r) {
    float v = c[r] + bb;
    if (tanh_act) v = tanhf(v);
    tile[(mb + r) * ld + n] = (_Float16)v;
  }
}

__device__ inline float sigmoidf_(float x) { return 1.0f / (1.0f + expf(-x)); }
__device__ inline void atomicAddF(float* p, float v) {
  __hip_atomic_fetch_add(p, v, __ATOMIC_RELAXED, __HIP_MEMORY_SCOPE_AGENT);
}

// ---------------- counts (mean-aggregation denominators) ----------------
__global__ void __launch_bounds__(256) gnn_counts_init(float* counts) {
  int i = blockIdx.x * 256 + threadIdx.x;
  if (i < N_NODES) counts[i] = 1.0f;  // self loop
}
__global__ void __launch_bounds__(256) gnn_counts_edges(const int* __restrict__ e_dst, float* counts) {
  int i = blockIdx.x * 256 + threadIdx.x;
  if (i < N_EDGES) atomicAddF(&counts[e_dst[i]], 1.0f);
}

// ---------------- pack per-step message-MLP weights -> f16 fragment layout ----------------
// Per step block (10240 halfs): [0,4096) W1 ncols=64; [4096,8192) W2 ncols=64; [8192,10240) W3 ncols=32
__global__ void __launch_bounds__(256) gnn_pack_weights(const float* __restrict__ W1,
                                                        const float* __restrict__ W2,
                                                        const float* __restrict__ W3,
                                                        _Float16* __restrict__ wp) {
  int i = blockIdx.x * 256 + threadIdx.x;
  if (i >= STEPS * 10240) return;
  int l = i / 10240, r = i % 10240;
  _Float16* dst = wp + l * 10240;
  if (r < 4096) {
    int k = r >> 6, n = r & 63;
    dst[((k >> 4) * 64 + n) * 16 + (k & 15)] = (_Float16)W1[l * 4096 + r];
  } else if (r < 8192) {
    int s = r - 4096, k = s >> 6, n = s & 63;
    dst[4096 + ((k >> 4) * 64 + n) * 16 + (k & 15)] = (_Float16)W2[l * 4096 + s];
  } else {
    int s = r - 8192, k = s >> 5, n = s & 31;
    dst[8192 + ((k >> 4) * 32 + n) * 16 + (k & 15)] = (_Float16)W3[l * 2048 + s];
  }
}

// ---------------- encoder: h = relu(x @ enc_W + b) ----------------
__global__ void __launch_bounds__(256) gnn_encoder(const float* __restrict__ x,
                                                   const float* __restrict__ W,   // [64][32]
                                                   const float* __restrict__ b,
                                                   _Float16* __restrict__ h16) {
  __shared__ _Float16 ws[IN_DIM * HID];   // fragment-packed, ncols=32
  int tid = threadIdx.x;
  for (int s = tid; s < IN_DIM * HID; s += 256) {
    int k = s >> 5, n = s & 31;
    ws[((k >> 4) * 32 + n) * 16 + (k & 15)] = (_Float16)W[s];
  }
  __syncthreads();

  int wave = tid >> 5, lane = tid & 31;
  int lm = lane & 15, kh = (lane >> 4) << 3, mb = (lane >> 4) << 3;
  float bv[2];
#pragma unroll
  for (int nt = 0; nt < 2; ++nt) bv[nt] = b[nt * 16 + lm];

  const int ntiles = N_NODES >> 4;     // exact
  for (int tile = blockIdx.x * NW + wave; tile < ntiles; tile += gridDim.x * NW) {
    int rb = tile << 4;
    const float* xr = x + (size_t)(rb + lm) * IN_DIM;
    v16h a0, a1;
#pragma unroll
    for (int i = 0; i < 8; ++i) {
      a0[i]     = (_Float16)xr[kh + i];
      a0[8 + i] = (_Float16)xr[16 + kh + i];
      a1[i]     = (_Float16)xr[32 + kh + i];
      a1[8 + i] = (_Float16)xr[48 + kh + i];
    }
#pragma unroll
    for (int nt = 0; nt < 2; ++nt) {
      v8f c = wmma_k64(a0, a1, ws, HID, nt * 16, lane);
      int n = nt * 16 + lm;
#pragma unroll
      for (int r8 = 0; r8 < 8; ++r8)
        h16[(size_t)(rb + mb + r8) * HID + n] = (_Float16)fmaxf(c[r8] + bv[nt], 0.0f);
    }
  }
}

// ---------------- per-edge message MLP + scatter-add into agg ----------------
__global__ void __launch_bounds__(256) gnn_edge_mlp(const _Float16* __restrict__ h16,
                                                    const int* __restrict__ e_src,
                                                    const int* __restrict__ e_dst,
                                                    const _Float16* __restrict__ wp,  // 10240 halfs packed
                                                    const float* __restrict__ b1,
                                                    const float* __restrict__ b2,
                                                    const float* __restrict__ b3,
                                                    float* __restrict__ agg) {
  __shared__ _Float16 wl[10240];          // W1|W2|W3 fragment-packed (20480B)
  __shared__ _Float16 bufs[NW][16 * LDT]; // inter-layer tile per wave

  int tid = threadIdx.x, wave = tid >> 5, lane = tid & 31;

  if (wave == 0) {
    // Tensor Data Mover: DMA the 20480B packed-weight block into LDS (1D tile)
    uint64_t ga = (uint64_t)(uintptr_t)wp;
    uint32_t la = (uint32_t)(uintptr_t)&wl[0];
    const int nd8 = 20480 / 8;            // data_size=3 (8B units)
    u32x4 g0;
    g0[0] = 1u;                                              // count=1 (valid descriptor)
    g0[1] = la;                                              // lds_addr
    g0[2] = (uint32_t)ga;                                    // global_addr[31:0]
    g0[3] = (uint32_t)((ga >> 32) & 0x1FFFFFFu) | (2u << 30);// global_addr[56:32] | type=2
    i32x8 g1;
    g1[0] = 3 << 16;                    // data_size=3 (8B), no multicast/flags
    g1[1] = (nd8 & 0xFFFF) << 16;       // tensor_dim0[15:0]
    g1[2] = (nd8 >> 16) | (1 << 16);    // tensor_dim0[31:16], tensor_dim1=1
    g1[3] = nd8 << 16;                  // tile_dim0
    g1[4] = 0;                          // tile_dim1/2 unused
    g1[5] = nd8;                        // tensor_dim0_stride
    g1[6] = 0;
    g1[7] = 0;
    i32x4 z4 = {0, 0, 0, 0};
    i32x8 z8 = {0, 0, 0, 0, 0, 0, 0, 0};
    __builtin_amdgcn_tensor_load_to_lds(g0, g1, z4, z4, z8, 0);
    __builtin_amdgcn_s_wait_tensorcnt(0);
  }
  __syncthreads();

  const _Float16* w1p = wl;
  const _Float16* w2p = wl + 4096;
  const _Float16* w3p = wl + 8192;
  _Float16* buf = bufs[wave];

  int lm = lane & 15, kh = (lane >> 4) << 3, mb = (lane >> 4) << 3;
  float b1v[4], b2v[4], b3v[2];
#pragma unroll
  for (int nt = 0; nt < 4; ++nt) { b1v[nt] = b1[nt * 16 + lm]; b2v[nt] = b2[nt * 16 + lm]; }
#pragma unroll
  for (int nt = 0; nt < 2; ++nt) b3v[nt] = b3[nt * 16 + lm];

  const int ntiles = (N_EDGES + N_NODES) >> 4;   // exact: 53125
  for (int tile = blockIdx.x * NW + wave; tile < ntiles; tile += gridDim.x * NW) {
    int rb = tile << 4;
    // A fragments straight from global h16 (row = concat(h[dst], h[src]))
    int row  = rb + lm;
    int ridx = row < N_EDGES ? row : 0;
    int nd   = row < N_EDGES ? e_dst[ridx] : row - N_EDGES;
    int ns   = row < N_EDGES ? e_src[ridx] : row - N_EDGES;
    v16h a0 = ld_a_gl(h16 + (size_t)nd * HID, kh);   // cols 0..31  = h[dst]
    v16h a1 = ld_a_gl(h16 + (size_t)ns * HID, kh);   // cols 32..63 = h[src]

    // destination nodes for the scatter (rows mb..mb+7 of this tile)
    int dnv[8];
#pragma unroll
    for (int r8 = 0; r8 < 8; ++r8) {
      int g  = rb + mb + r8;
      int gc = g < N_EDGES ? g : 0;
      int v  = e_dst[gc];
      dnv[r8] = g < N_EDGES ? v : g - N_EDGES;
    }

    __builtin_amdgcn_wave_barrier();
    // layer 1: 64 -> 64, tanh
#pragma unroll
    for (int nt = 0; nt < 4; ++nt) {
      v8f c = wmma_k64(a0, a1, w1p, 64, nt * 16, lane);
      st_c_f16(buf, LDT, nt * 16, lane, c, b1v[nt], true);
    }
    __builtin_amdgcn_wave_barrier();
    a0 = ld_a(buf, LDT, 0, lane); a1 = ld_a(buf, LDT, 32, lane);
    __builtin_amdgcn_wave_barrier();
    // layer 2: 64 -> 64, tanh
#pragma unroll
    for (int nt = 0; nt < 4; ++nt) {
      v8f c = wmma_k64(a0, a1, w2p, 64, nt * 16, lane);
      st_c_f16(buf, LDT, nt * 16, lane, c, b2v[nt], true);
    }
    __builtin_amdgcn_wave_barrier();
    a0 = ld_a(buf, LDT, 0, lane); a1 = ld_a(buf, LDT, 32, lane);
    __builtin_amdgcn_wave_barrier();
    // layer 3: 64 -> 32; scatter-add into agg[dst]
#pragma unroll
    for (int nt = 0; nt < 2; ++nt) {
      v8f c = wmma_k64(a0, a1, w3p, 32, nt * 16, lane);
      int n = nt * 16 + lm;
#pragma unroll
      for (int r8 = 0; r8 < 8; ++r8)
        atomicAddF(&agg[(size_t)dnv[r8] * HID + n], c[r8] + b3v[nt]);
    }
    __builtin_amdgcn_wave_barrier();
  }
}

// ---------------- GRU cell update (in place on h16) ----------------
__global__ void __launch_bounds__(256) gnn_gru(_Float16* __restrict__ h16,
                                               const float* __restrict__ agg,
                                               const float* __restrict__ counts,
                                               const float* __restrict__ Wih,  // [96][32] torch
                                               const float* __restrict__ Whh,
                                               const float* __restrict__ bih,
                                               const float* __restrict__ bhh) {
  __shared__ _Float16 wih_s[2 * 96 * 16];  // fragment-packed B = Wih^T (ncols=96)
  __shared__ _Float16 whh_s[2 * 96 * 16];
  int tid = threadIdx.x;
  for (int i = tid; i < 96 * HID; i += 256) {   // i = n*32 + k (source linear)
    int k = i & 31, n = i >> 5;
    int d = ((k >> 4) * 96 + n) * 16 + (k & 15);
    wih_s[d] = (_Float16)Wih[i];
    whh_s[d] = (_Float16)Whh[i];
  }
  __syncthreads();

  int wave = tid >> 5, lane = tid & 31;
  int lm = lane & 15, kh = (lane >> 4) << 3, mb = (lane >> 4) << 3;
  float biv[6], bhv[6];
#pragma unroll
  for (int g = 0; g < 6; ++g) { biv[g] = bih[g * 16 + lm]; bhv[g] = bhh[g * 16 + lm]; }

  const int ntiles = N_NODES >> 4;
  for (int tile = blockIdx.x * NW + wave; tile < ntiles; tile += gridDim.x * NW) {
    int rb = tile << 4;
    int node = rb + lm;
    float ic = 1.0f / counts[node];
    const float* ar = agg + (size_t)node * HID;
    v16h aA;
#pragma unroll
    for (int i = 0; i < 8; ++i) {
      aA[i]     = (_Float16)(ar[kh + i] * ic);
      aA[8 + i] = (_Float16)(ar[16 + kh + i] * ic);
    }
    v16h aH = ld_a_gl(h16 + (size_t)node * HID, kh);

    v8f gi[6], gh[6];
#pragma unroll
    for (int g = 0; g < 6; ++g) {
      gi[g] = wmma_k32(aA, wih_s, 96, g * 16, lane);
      gh[g] = wmma_k32(aH, whh_s, 96, g * 16, lane);
    }
#pragma unroll
    for (int j = 0; j < 2; ++j) {        // two 16-col halves of HID
      int col = j * 16 + lm;
#pragma unroll
      for (int r8 = 0; r8 < 8; ++r8) {
        float rg = sigmoidf_((gi[0 + j][r8] + biv[0 + j]) + (gh[0 + j][r8] + bhv[0 + j]));
        float zg = sigmoidf_((gi[2 + j][r8] + biv[2 + j]) + (gh[2 + j][r8] + bhv[2 + j]));
        float ng = tanhf((gi[4 + j][r8] + biv[4 + j]) + rg * (gh[4 + j][r8] + bhv[4 + j]));
        size_t idx = (size_t)(rb + mb + r8) * HID + col;
        float hold = (float)h16[idx];
        h16[idx] = (_Float16)((1.0f - zg) * ng + zg * hold);
      }
    }
  }
}

// ---------------- decoder: tanh(tanh(h W1+b1) W2+b2) W3+b3 ----------------
__global__ void __launch_bounds__(256) gnn_decoder(const _Float16* __restrict__ h16,
                                                   const float* __restrict__ W1, const float* __restrict__ b1, // [32][64]
                                                   const float* __restrict__ W2, const float* __restrict__ b2, // [64][64]
                                                   const float* __restrict__ W3, const float* __restrict__ b3, // [64][8]
                                                   float* __restrict__ out) {
  __shared__ _Float16 w1s[2 * 64 * 16];   // K=32, ncols=64
  __shared__ _Float16 w2s[4 * 64 * 16];   // K=64, ncols=64
  __shared__ _Float16 w3s[4 * 16 * 16];   // K=64, ncols=16 (cols 8..15 zero)
  __shared__ _Float16 dbuf[NW][16 * LDT];

  int tid = threadIdx.x;
  for (int s = tid; s < HID * MLP_HID; s += 256) {         // W1: k<32, n<64
    int k = s >> 6, n = s & 63;
    w1s[((k >> 4) * 64 + n) * 16 + (k & 15)] = (_Float16)W1[s];
  }
  for (int s = tid; s < MLP_HID * MLP_HID; s += 256) {     // W2
    int k = s >> 6, n = s & 63;
    w2s[((k >> 4) * 64 + n) * 16 + (k & 15)] = (_Float16)W2[s];
  }
  for (int s = tid; s < MLP_HID * 16; s += 256) {          // W3 padded to 16 cols
    int k = s >> 4, n = s & 15;
    w3s[((k >> 4) * 16 + n) * 16 + (k & 15)] = (n < OUT_DIM) ? (_Float16)W3[k * OUT_DIM + n] : (_Float16)0.0f;
  }
  __syncthreads();

  int wave = tid >> 5, lane = tid & 31;
  int lm = lane & 15, kh = (lane >> 4) << 3, mb = (lane >> 4) << 3;
  _Float16* buf = dbuf[wave];
  float b1v[4], b2v[4];
#pragma unroll
  for (int nt = 0; nt < 4; ++nt) { b1v[nt] = b1[nt * 16 + lm]; b2v[nt] = b2[nt * 16 + lm]; }
  float b3v = (lm < OUT_DIM) ? b3[lm] : 0.0f;

  const int ntiles = N_NODES >> 4;
  for (int tile = blockIdx.x * NW + wave; tile < ntiles; tile += gridDim.x * NW) {
    int rb = tile << 4;
    v16h aH = ld_a_gl(h16 + (size_t)(rb + lm) * HID, kh);

    __builtin_amdgcn_wave_barrier();
#pragma unroll
    for (int nt = 0; nt < 4; ++nt) {
      v8f c = wmma_k32(aH, w1s, 64, nt * 16, lane);
      st_c_f16(buf, LDT, nt * 16, lane, c, b1v[nt], true);
    }
    __builtin_amdgcn_wave_barrier();
    v16h a0 = ld_a(buf, LDT, 0, lane), a1 = ld_a(buf, LDT, 32, lane);
    __builtin_amdgcn_wave_barrier();
#pragma unroll
    for (int nt = 0; nt < 4; ++nt) {
      v8f c = wmma_k64(a0, a1, w2s, 64, nt * 16, lane);
      st_c_f16(buf, LDT, nt * 16, lane, c, b2v[nt], true);
    }
    __builtin_amdgcn_wave_barrier();
    a0 = ld_a(buf, LDT, 0, lane); a1 = ld_a(buf, LDT, 32, lane);
    __builtin_amdgcn_wave_barrier();
    {
      v8f c = wmma_k64(a0, a1, w3s, 16, 0, lane);
      if (lm < OUT_DIM) {
#pragma unroll
        for (int r8 = 0; r8 < 8; ++r8)
          out[(size_t)(rb + mb + r8) * OUT_DIM + lm] = c[r8] + b3v;
      }
    }
  }
}

// ---------------- host launcher ----------------
extern "C" void kernel_launch(void* const* d_in, const int* in_sizes, int n_in,
                              void* d_out, int out_size, void* d_ws, size_t ws_size,
                              hipStream_t stream) {
  (void)in_sizes; (void)n_in; (void)out_size; (void)ws_size;
  const float* x      = (const float*)d_in[0];
  const int*   eidx   = (const int*)d_in[1];
  const int*   e_src  = eidx;             // edge_index[0] (x_j source)
  const int*   e_dst  = eidx + N_EDGES;   // edge_index[1] (x_i / segment)
  const float* enc_W  = (const float*)d_in[2];
  const float* enc_b  = (const float*)d_in[3];
  const float* msg_W1 = (const float*)d_in[4];
  const float* msg_b1 = (const float*)d_in[5];
  const float* msg_W2 = (const float*)d_in[6];
  const float* msg_b2 = (const float*)d_in[7];
  const float* msg_W3 = (const float*)d_in[8];
  const float* msg_b3 = (const float*)d_in[9];
  const float* gru_Wih = (const float*)d_in[10];
  const float* gru_Whh = (const float*)d_in[11];
  const float* gru_bih = (const float*)d_in[12];
  const float* gru_bhh = (const float*)d_in[13];
  const float* dec_W1 = (const float*)d_in[14];
  const float* dec_b1 = (const float*)d_in[15];
  const float* dec_W2 = (const float*)d_in[16];
  const float* dec_b2 = (const float*)d_in[17];
  const float* dec_W3 = (const float*)d_in[18];
  const float* dec_b3 = (const float*)d_in[19];
  float* out = (float*)d_out;

  size_t off = 0;
  auto take = [&](size_t bytes) -> void* {
    void* p = (char*)d_ws + off;
    off += (bytes + 255) & ~(size_t)255;
    return p;
  };
  _Float16* h16    = (_Float16*)take((size_t)N_NODES * HID * sizeof(_Float16));
  float*    agg    = (float*)take((size_t)N_NODES * HID * sizeof(float));
  float*    counts = (float*)take((size_t)N_NODES * sizeof(float));
  _Float16* wpack  = (_Float16*)take((size_t)STEPS * 10240 * sizeof(_Float16));

  const int nodeBlocks = ((N_NODES >> 4) + NW - 1) / NW;   // 391
  const int edgeBlocks = 2048;                             // grid-stride over 53125 tiles

  gnn_counts_init<<<(N_NODES + 255) / 256, 256, 0, stream>>>(counts);
  gnn_counts_edges<<<(N_EDGES + 255) / 256, 256, 0, stream>>>(e_dst, counts);
  gnn_pack_weights<<<(STEPS * 10240 + 255) / 256, 256, 0, stream>>>(msg_W1, msg_W2, msg_W3, wpack);
  gnn_encoder<<<nodeBlocks, 256, 0, stream>>>(x, enc_W, enc_b, h16);

  for (int l = 0; l < STEPS; ++l) {
    (void)hipMemsetAsync(agg, 0, (size_t)N_NODES * HID * sizeof(float), stream);
    gnn_edge_mlp<<<edgeBlocks, 256, 0, stream>>>(
        h16, e_src, e_dst, wpack + (size_t)l * 10240,
        msg_b1 + (size_t)l * MLP_HID, msg_b2 + (size_t)l * MLP_HID, msg_b3 + (size_t)l * HID,
        agg);
    gnn_gru<<<nodeBlocks, 256, 0, stream>>>(
        h16, agg, counts,
        gru_Wih + (size_t)l * 3 * HID * HID, gru_Whh + (size_t)l * 3 * HID * HID,
        gru_bih + (size_t)l * 3 * HID,       gru_bhh + (size_t)l * 3 * HID);
  }

  gnn_decoder<<<nodeBlocks, 256, 0, stream>>>(h16, dec_W1, dec_b1, dec_W2, dec_b2,
                                              dec_W3, dec_b3, out);
}
